// RNNBlock_29188597744120
// MI455X (gfx1250) — matrix operand order repaired and checked by reading, not compiled, not run on hardware
//
#include <hip/hip_runtime.h>
#include <hip/hip_bf16.h>

typedef __attribute__((ext_vector_type(16))) __bf16 v16bf;
typedef __attribute__((ext_vector_type(8)))  __bf16 v8bf;
typedef __attribute__((ext_vector_type(8)))  float  v8f;

#define WMMA_BF16(a, b, c) \
  __builtin_amdgcn_wmma_f32_16x16x32_bf16(false, (a), false, (b), (short)0, (c), false, false)

static constexpr int Tn = 4096;
static constexpr int Dn = 2048;
static constexpr int Hn = 32;
static constexpr int DHn = 64;

__device__ inline v8f vzero8() {
  v8f z;
  for (int i = 0; i < 8; ++i) z[i] = 0.f;
  return z;
}

// LDS byte offset of a __shared__ pointer: flat LDS aperture keeps the LDS
// offset in addr[31:0] (CDNA5 ISA 10.2), so truncation is the offset.
__device__ inline unsigned lds_addr(const void* p) {
  return (unsigned)(unsigned long long)p;
}

// Issue two async 16B global->LDS copies (GVS mode: SGPR64 base + VGPR32 offset).
__device__ inline void async_copy_32B(unsigned lds_b, unsigned glb_b, const void* base) {
  asm volatile(
      "global_load_async_to_lds_b128 %0, %2, %4\n\t"
      "global_load_async_to_lds_b128 %1, %3, %4"
      :
      : "v"(lds_b), "v"(lds_b + 16), "v"(glb_b), "v"(glb_b + 16),
        "s"((unsigned long long)base)
      : "memory");
}

__device__ inline void async_wait() {
  asm volatile("s_wait_asynccnt 0" ::: "memory");
}

// ---- WMMA fragment loaders (wave32; element i of lane half h is k = 16h + i in A and B) ----------
__device__ inline v16bf a_frag(const __bf16* s, int ld, int rowBase, int kBase, int lane) {
  int r = lane & 15;
  int koff = (lane & 16) ? 16 : 0;
  const __bf16* p = s + (rowBase + r) * ld + kBase + koff;
  v8bf lo = *(const v8bf*)p;
  v8bf hi = *(const v8bf*)(p + 8);
  v16bf a;
  for (int i = 0; i < 8; ++i) { a[i] = lo[i]; a[i + 8] = hi[i]; }
  return a;
}

__device__ inline v16bf b_frag(const __bf16* st, int ld, int colBase, int kBase, int lane) {
  int n = lane & 15;
  int krow = (lane & 16) ? 16 : 0;
  const __bf16* p = st + (colBase + n) * ld + kBase + krow;
  v8bf lo = *(const v8bf*)p;
  v8bf hi = *(const v8bf*)(p + 8);
  v16bf b;
  for (int i = 0; i < 8; ++i) { b[i] = lo[i]; b[i + 8] = hi[i]; }
  return b;
}

__device__ inline v16bf b_frag_f32(const float* st, int ld, int colBase, int kBase, int lane) {
  int n = lane & 15;
  int krow = (lane & 16) ? 16 : 0;
  const float* p = st + (colBase + n) * ld + kBase + krow;
  v16bf b;
  for (int i = 0; i < 16; ++i) b[i] = (__bf16)p[i];
  return b;
}

// ---- fp32 -> bf16 conversion --------------------------------------------
__global__ __launch_bounds__(256) void cvt_bf16_kernel(const float* __restrict__ in,
                                                       __bf16* __restrict__ out, int n) {
  for (int i = blockIdx.x * 256 + threadIdx.x; i < n; i += gridDim.x * 256)
    out[i] = (__bf16)in[i];
}

// ---- LayerNorm 1: xn = LN(x) as bf16 ------------------------------------
__global__ __launch_bounds__(256) void ln1_kernel(const float* __restrict__ x,
                                                  const float* __restrict__ w,
                                                  const float* __restrict__ b,
                                                  __bf16* __restrict__ xn) {
  __shared__ float rs[256], rq[256];
  int row = blockIdx.x, tid = threadIdx.x;
  const float* xr = x + (size_t)row * Dn;
  float v8[8], s = 0.f, q = 0.f;
  for (int i = 0; i < 8; ++i) {
    float v = xr[tid + i * 256];
    v8[i] = v; s += v; q += v * v;
  }
  rs[tid] = s; rq[tid] = q; __syncthreads();
  for (int off = 128; off > 0; off >>= 1) {
    if (tid < off) { rs[tid] += rs[tid + off]; rq[tid] += rq[tid + off]; }
    __syncthreads();
  }
  float mean = rs[0] / Dn;
  float var = rq[0] / Dn - mean * mean;
  float rstd = rsqrtf(var + 1e-5f);
  for (int i = 0; i < 8; ++i) {
    int c = tid + i * 256;
    xn[(size_t)row * Dn + c] = (__bf16)((v8[i] - mean) * rstd * w[c] + b[c]);
  }
}

// ---- LayerNorm 2 + residual: u = x + o (in place over O); x2 = LN(u) -----
__global__ __launch_bounds__(256) void ln2_kernel(const float* __restrict__ x,
                                                  float* __restrict__ O,   // in: o, out: u
                                                  const float* __restrict__ w,
                                                  const float* __restrict__ b,
                                                  __bf16* __restrict__ x2) {
  __shared__ float rs[256], rq[256];
  int row = blockIdx.x, tid = threadIdx.x;
  const float* xr = x + (size_t)row * Dn;
  float* orow = O + (size_t)row * Dn;
  float v8[8], s = 0.f, q = 0.f;
  for (int i = 0; i < 8; ++i) {
    int c = tid + i * 256;
    float u = xr[c] + orow[c];
    v8[i] = u; s += u; q += u * u;
  }
  rs[tid] = s; rq[tid] = q; __syncthreads();
  for (int off = 128; off > 0; off >>= 1) {
    if (tid < off) { rs[tid] += rs[tid + off]; rq[tid] += rq[tid + off]; }
    __syncthreads();
  }
  float mean = rs[0] / Dn;
  float var = rq[0] / Dn - mean * mean;
  float rstd = rsqrtf(var + 1e-5f);
  for (int i = 0; i < 8; ++i) {
    int c = tid + i * 256;
    orow[c] = v8[i];  // u = x + o
    x2[(size_t)row * Dn + c] = (__bf16)((v8[i] - mean) * rstd * w[c] + b[c]);
  }
}

// ---- SiLU(mlp_in) * sigmoid(gate) ----------------------------------------
__global__ __launch_bounds__(256) void act_kernel(const __bf16* __restrict__ G,
                                                  __bf16* __restrict__ M) {
  size_t total = (size_t)Tn * Dn;
  for (size_t i = blockIdx.x * 256ull + threadIdx.x; i < total; i += (size_t)gridDim.x * 256) {
    size_t t = i / Dn, d = i % Dn;
    float g = (float)G[t * (2 * Dn) + d];
    float m = (float)G[t * (2 * Dn) + Dn + d];
    float sg = 1.f / (1.f + expf(-g));
    float sm = 1.f / (1.f + expf(-m));
    M[i] = (__bf16)(m * sm * sg);
  }
}

// ---- bf16 GEMM: C[M,N] = A[M,K] @ B[K,N], 128x64 block tile, BK=64 --------
// A tile staged with global_load_async_to_lds_b128 (ASYNCcnt path);
// B tile transposed through VGPRs. MODE 0: bf16 C. MODE 2: fp32 C = acc + Add.
template <int MODE>
__global__ __launch_bounds__(256) void gemm_bf16_kernel(const __bf16* __restrict__ A,
                                                        const __bf16* __restrict__ B,
                                                        __bf16* __restrict__ Cbf,
                                                        float* __restrict__ Cf,
                                                        const float* __restrict__ Add,
                                                        int M, int N, int K) {
  constexpr int LDA = 72;  // 144B rows: 16B-aligned, bank-conflict skewed
  __shared__ __align__(16) __bf16 As[128 * LDA];
  __shared__ __align__(16) __bf16 Bt[64 * LDA];
  int bm = blockIdx.y * 128, bn = blockIdx.x * 64;
  int tid = threadIdx.x, lane = tid & 31, w = tid >> 5;
  int wrow = w >> 1, wcol = w & 1;  // waves: 4 (M) x 2 (N), each owns 32x32
  v8f acc[2][2];
  for (int i = 0; i < 2; ++i)
    for (int j = 0; j < 2; ++j) acc[i][j] = vzero8();

  // per-thread staging coordinates
  int arow = tid >> 1, aseg = tid & 1;                    // A: 2 thr/row, 32 els each
  unsigned albase = lds_addr(&As[arow * LDA + aseg * 32]);
  int bk = tid & 63, bg = tid >> 6;                       // B: 64 k x 4 groups of 16 n

  for (int k0 = 0; k0 < K; k0 += 64) {
    // async-stage A tile 128x64 (4 x b128 per thread, ASYNCcnt-tracked)
    unsigned agb = (unsigned)(((size_t)(bm + arow) * K + k0 + aseg * 32) * 2);
    async_copy_32B(albase, agb, A);
    async_copy_32B(albase + 32, agb + 32, A);
    // stage B tile 64x64 transposed -> Bt[n][k]
    {
      const __bf16* gp = B + (size_t)(k0 + bk) * N + bn + bg * 16;
      v8bf x0 = *(const v8bf*)gp;
      v8bf x1 = *(const v8bf*)(gp + 8);
      for (int i = 0; i < 8; ++i) {
        Bt[(bg * 16 + i) * LDA + bk] = x0[i];
        Bt[(bg * 16 + 8 + i) * LDA + bk] = x1[i];
      }
      if (k0 + 64 < K) __builtin_prefetch(gp + 64 * (size_t)N, 0, 1);  // global_prefetch_b8
    }
    async_wait();        // this wave's async LDS writes done
    __syncthreads();     // all waves' tiles visible
    for (int kk = 0; kk < 64; kk += 32) {
      v16bf af[2], bfr[2];
      af[0] = a_frag(As, LDA, wrow * 32, kk, lane);
      af[1] = a_frag(As, LDA, wrow * 32 + 16, kk, lane);
      bfr[0] = b_frag(Bt, LDA, wcol * 32, kk, lane);
      bfr[1] = b_frag(Bt, LDA, wcol * 32 + 16, kk, lane);
      for (int mi = 0; mi < 2; ++mi)
        for (int ni = 0; ni < 2; ++ni)
          acc[mi][ni] = WMMA_BF16(af[mi], bfr[ni], acc[mi][ni]);
    }
    __syncthreads();
  }

  for (int mi = 0; mi < 2; ++mi)
    for (int ni = 0; ni < 2; ++ni)
      for (int r = 0; r < 8; ++r) {
        int row = bm + wrow * 32 + mi * 16 + r + ((lane & 16) ? 8 : 0);
        int col = bn + wcol * 32 + ni * 16 + (lane & 15);
        float v = acc[mi][ni][r];
        if constexpr (MODE == 0)
          Cbf[(size_t)row * N + col] = (__bf16)v;
        else
          Cf[(size_t)row * N + col] = v + Add[(size_t)row * N + col];
      }
}

// ---- chunked causal linear attention, one block per head -----------------
// Carries S[i][j] = sum_{tau} k[i] v[j] (fp32, LDS). gamma handled analytically.
__global__ __launch_bounds__(256) void attn_kernel(const __bf16* __restrict__ Q,
                                                   const __bf16* __restrict__ K,
                                                   const __bf16* __restrict__ V,
                                                   const float* __restrict__ gamma,
                                                   const float* __restrict__ state_in,
                                                   float* __restrict__ O,
                                                   float* __restrict__ state_out) {
  constexpr int LD = 72;   // bf16 tile leading dim
  constexpr int SLD = 68;  // fp32 S leading dim
  __shared__ __align__(16) __bf16 Qs[64 * LD], Vs[64 * LD], Vt[64 * LD], Kt[64 * LD], Pb[64 * LD];
  __shared__ float S[64 * SLD];
  __shared__ float sumq[64], gam[64];

  int h = blockIdx.x;
  int tid = threadIdx.x, lane = tid & 31, w = tid >> 5;
  int wrow = w >> 1, wcol = w & 1;
  int hb = h * DHn;

  for (int i = tid; i < DHn * DHn; i += 256) {
    int r = i >> 6, c = i & 63;
    S[r * SLD + c] = state_in[(size_t)h * DHn * DHn + i];
  }
  if (tid < 64) gam[tid] = gamma[(size_t)h * DHn + tid];
  __syncthreads();

  int row = tid >> 2, seg = tid & 3;  // staging coords: 4 thr/row, 16 els each
  unsigned qlds = lds_addr(&Qs[row * LD + seg * 16]);

  for (int c0 = 0; c0 < Tn; c0 += 64) {
    __syncthreads();  // previous iteration's tile reads done before restaging
    {
      size_t goff = (size_t)(c0 + row) * Dn + hb + seg * 16;
      // Q tile: async global->LDS
      async_copy_32B(qlds, (unsigned)(goff * 2), Q);
      // V row-major + V/K transposed through VGPRs
      v8bf v0 = *(const v8bf*)(V + goff), v1 = *(const v8bf*)(V + goff + 8);
      v8bf k0 = *(const v8bf*)(K + goff), k1 = *(const v8bf*)(K + goff + 8);
      *(v8bf*)&Vs[row * LD + seg * 16] = v0;
      *(v8bf*)&Vs[row * LD + seg * 16 + 8] = v1;
      for (int i = 0; i < 8; ++i) {
        Kt[(seg * 16 + i) * LD + row] = k0[i];
        Kt[(seg * 16 + 8 + i) * LD + row] = k1[i];
        Vt[(seg * 16 + i) * LD + row] = v0[i];
        Vt[(seg * 16 + 8 + i) * LD + row] = v1[i];
      }
    }
    async_wait();
    __syncthreads();
    if (tid < 64) {  // row sums of Qc for the analytic gamma term
      float s = 0.f;
      for (int j = 0; j < 64; ++j) s += (float)Qs[tid * LD + j];
      sumq[tid] = s;
    }
    __syncthreads();

    // O_inter = Qc @ S_prev^T
    v8f acc[2][2];
    for (int i = 0; i < 2; ++i)
      for (int j = 0; j < 2; ++j) acc[i][j] = vzero8();
    for (int kk = 0; kk < 64; kk += 32) {
      v16bf af[2], bfr[2];
      af[0] = a_frag(Qs, LD, wrow * 32, kk, lane);
      af[1] = a_frag(Qs, LD, wrow * 32 + 16, kk, lane);
      bfr[0] = b_frag_f32(S, SLD, wcol * 32, kk, lane);
      bfr[1] = b_frag_f32(S, SLD, wcol * 32 + 16, kk, lane);
      for (int mi = 0; mi < 2; ++mi)
        for (int ni = 0; ni < 2; ++ni)
          acc[mi][ni] = WMMA_BF16(af[mi], bfr[ni], acc[mi][ni]);
    }

    // P = Qc @ Vc^T, causal mask (tau <= t), store bf16
    v8f pac[2][2];
    for (int i = 0; i < 2; ++i)
      for (int j = 0; j < 2; ++j) pac[i][j] = vzero8();
    for (int kk = 0; kk < 64; kk += 32) {
      v16bf af[2], bfr[2];
      af[0] = a_frag(Qs, LD, wrow * 32, kk, lane);
      af[1] = a_frag(Qs, LD, wrow * 32 + 16, kk, lane);
      bfr[0] = b_frag(Vs, LD, wcol * 32, kk, lane);
      bfr[1] = b_frag(Vs, LD, wcol * 32 + 16, kk, lane);
      for (int mi = 0; mi < 2; ++mi)
        for (int ni = 0; ni < 2; ++ni)
          pac[mi][ni] = WMMA_BF16(af[mi], bfr[ni], pac[mi][ni]);
    }
    for (int mi = 0; mi < 2; ++mi)
      for (int ni = 0; ni < 2; ++ni)
        for (int r = 0; r < 8; ++r) {
          int tr = wrow * 32 + mi * 16 + r + ((lane & 16) ? 8 : 0);
          int tc = wcol * 32 + ni * 16 + (lane & 15);
          Pb[tr * LD + tc] = (__bf16)((tc <= tr) ? pac[mi][ni][r] : 0.f);
        }
    __syncthreads();  // Pb visible; all O_inter reads of S are done

    // O_intra += P @ Kc ; accumulate into acc
    for (int kk = 0; kk < 64; kk += 32) {
      v16bf af[2], bfr[2];
      af[0] = a_frag(Pb, LD, wrow * 32, kk, lane);
      af[1] = a_frag(Pb, LD, wrow * 32 + 16, kk, lane);
      bfr[0] = b_frag(Kt, LD, wcol * 32, kk, lane);
      bfr[1] = b_frag(Kt, LD, wcol * 32 + 16, kk, lane);
      for (int mi = 0; mi < 2; ++mi)
        for (int ni = 0; ni < 2; ++ni)
          acc[mi][ni] = WMMA_BF16(af[mi], bfr[ni], acc[mi][ni]);
    }
    // write O with analytic gamma term: (t+1) * gamma[i] * sum_j q_t[j]
    for (int mi = 0; mi < 2; ++mi)
      for (int ni = 0; ni < 2; ++ni)
        for (int r = 0; r < 8; ++r) {
          int tr = wrow * 32 + mi * 16 + r + ((lane & 16) ? 8 : 0);
          int tc = wcol * 32 + ni * 16 + (lane & 15);
          float v = acc[mi][ni][r] + (float)(c0 + tr + 1) * gam[tc] * sumq[tr];
          O[(size_t)(c0 + tr) * Dn + hb + tc] = v;
        }

    // S += Kc^T @ Vc
    v8f ds[2][2];
    for (int i = 0; i < 2; ++i)
      for (int j = 0; j < 2; ++j) ds[i][j] = vzero8();
    for (int kk = 0; kk < 64; kk += 32) {
      v16bf af[2], bfr[2];
      af[0] = a_frag(Kt, LD, wrow * 32, kk, lane);
      af[1] = a_frag(Kt, LD, wrow * 32 + 16, kk, lane);
      bfr[0] = b_frag(Vt, LD, wcol * 32, kk, lane);
      bfr[1] = b_frag(Vt, LD, wcol * 32 + 16, kk, lane);
      for (int mi = 0; mi < 2; ++mi)
        for (int ni = 0; ni < 2; ++ni)
          ds[mi][ni] = WMMA_BF16(af[mi], bfr[ni], ds[mi][ni]);
    }
    for (int mi = 0; mi < 2; ++mi)
      for (int ni = 0; ni < 2; ++ni)
        for (int r = 0; r < 8; ++r) {
          int sr = wrow * 32 + mi * 16 + r + ((lane & 16) ? 8 : 0);
          int sc = wcol * 32 + ni * 16 + (lane & 15);
          S[sr * SLD + sc] += ds[mi][ni][r];
        }
  }
  __syncthreads();
  // final state = S + T * gamma[i]
  for (int i = tid; i < DHn * DHn; i += 256) {
    int r = i >> 6, c = i & 63;
    state_out[(size_t)h * DHn * DHn + i] = S[r * SLD + c] + (float)Tn * gam[r];
  }
}

extern "C" void kernel_launch(void* const* d_in, const int* in_sizes, int n_in,
                              void* d_out, int out_size, void* d_ws, size_t ws_size,
                              hipStream_t stream) {
  const float* x = (const float*)d_in[0];
  const float* state = (const float*)d_in[1];
  const float* Wq = (const float*)d_in[2];
  const float* Wk = (const float*)d_in[3];
  const float* Wv = (const float*)d_in[4];
  const float* gamma = (const float*)d_in[5];
  const float* Wgate = (const float*)d_in[6];
  const float* Wdown = (const float*)d_in[7];
  const float* ln1_w = (const float*)d_in[8];
  const float* ln1_b = (const float*)d_in[9];
  const float* ln2_w = (const float*)d_in[10];
  const float* ln2_b = (const float*)d_in[11];
  float* out = (float*)d_out;                 // [T*D] outputs
  float* state_out = out + (size_t)Tn * Dn;   // [H*DH*DH] final state

  // ---- workspace layout ----
  char* ws = (char*)d_ws;
  size_t off = 0;
  auto carve = [&](size_t bytes) { char* p = ws + off; off += (bytes + 255) & ~size_t(255); return p; };
  __bf16* XN  = (__bf16*)carve((size_t)Tn * Dn * 2);       // xn bf16; reused as X2
  __bf16* Wqb = (__bf16*)carve((size_t)Dn * Dn * 2);
  __bf16* Wkb = (__bf16*)carve((size_t)Dn * Dn * 2);
  __bf16* Wvb = (__bf16*)carve((size_t)Dn * Dn * 2);
  __bf16* Wgb = (__bf16*)carve((size_t)Dn * 2 * Dn * 2);
  __bf16* Wdb = (__bf16*)carve((size_t)Dn * Dn * 2);
  __bf16* Qb  = (__bf16*)carve((size_t)Tn * Dn * 2);       // reused as Mact
  __bf16* Kb  = (__bf16*)carve((size_t)Tn * Dn * 2);
  __bf16* Vb  = (__bf16*)carve((size_t)Tn * Dn * 2);
  float*  Ob  = (float*)carve((size_t)Tn * Dn * 4);        // o, then u = x+o
  __bf16* Gb  = (__bf16*)carve((size_t)Tn * 2 * Dn * 2);   // gate GEMM output
  (void)ws_size; (void)in_sizes; (void)n_in; (void)out_size;

  // 1) weights -> bf16 (48 MB total: L2-resident on 192MB L2)
  cvt_bf16_kernel<<<4096, 256, 0, stream>>>(Wq, Wqb, Dn * Dn);
  cvt_bf16_kernel<<<4096, 256, 0, stream>>>(Wk, Wkb, Dn * Dn);
  cvt_bf16_kernel<<<4096, 256, 0, stream>>>(Wv, Wvb, Dn * Dn);
  cvt_bf16_kernel<<<8192, 256, 0, stream>>>(Wgate, Wgb, Dn * 2 * Dn);
  cvt_bf16_kernel<<<4096, 256, 0, stream>>>(Wdown, Wdb, Dn * Dn);

  // 2) LN1
  ln1_kernel<<<Tn, 256, 0, stream>>>(x, ln1_w, ln1_b, XN);

  // 3) Q/K/V GEMMs (4096x2048x2048, bf16 out)
  dim3 gqkv(Dn / 64, Tn / 128);
  gemm_bf16_kernel<0><<<gqkv, 256, 0, stream>>>(XN, Wqb, Qb, nullptr, nullptr, Tn, Dn, Dn);
  gemm_bf16_kernel<0><<<gqkv, 256, 0, stream>>>(XN, Wkb, Kb, nullptr, nullptr, Tn, Dn, Dn);
  gemm_bf16_kernel<0><<<gqkv, 256, 0, stream>>>(XN, Wvb, Vb, nullptr, nullptr, Tn, Dn, Dn);

  // 4) chunked causal linear attention (one block per head)
  attn_kernel<<<Hn, 256, 0, stream>>>(Qb, Kb, Vb, gamma, state, Ob, state_out);

  // 5) LN2 + residual (Ob becomes u = x + o; XN becomes x2)
  ln2_kernel<<<Tn, 256, 0, stream>>>(x, Ob, ln2_w, ln2_b, XN);

  // 6) gate GEMM (4096x4096x2048, bf16 out)
  dim3 ggate(2 * Dn / 64, Tn / 128);
  gemm_bf16_kernel<0><<<ggate, 256, 0, stream>>>(XN, Wgb, Gb, nullptr, nullptr, Tn, 2 * Dn, Dn);

  // 7) SiLU(mlp_in) * sigmoid(gate) -> Mact (reuses Qb)
  act_kernel<<<8192, 256, 0, stream>>>(Gb, Qb);

  // 8) down GEMM with fused residual: out = u + Mact @ Wdown
  gemm_bf16_kernel<2><<<gqkv, 256, 0, stream>>>(Qb, Wdb, nullptr, out, Ob, Tn, Dn, Dn);
}
